// SimpleModel_71811853189774
// MI455X (gfx1250) — matrix operand order repaired and checked by reading, not compiled, not run on hardware
//
#include <hip/hip_runtime.h>
#include <hip/hip_bf16.h>
#include <math.h>

// ---- problem constants (match reference) ----
constexpr int cB = 8, cS = 1024, cD = 512, cH = 8, cHD = 64;
constexpr int cBS = cB * cS;            // 8192 rows
constexpr size_t OUT0 = (size_t)cB * cS * cD;   // out elements before mask

typedef __attribute__((ext_vector_type(16))) __bf16 v16bf;
typedef __attribute__((ext_vector_type(8)))  __bf16 v8bf;
typedef __attribute__((ext_vector_type(8)))  float  v8f;

__device__ __forceinline__ __bf16 f2bf(float f) { return (__bf16)f; }

#define WMMA_BF16(A, Bm, C) \
  __builtin_amdgcn_wmma_f32_16x16x32_bf16(false, (A), false, (Bm), (short)0, (C), false, false)

// ---------------- workspace layout (bytes) ----------------
constexpr size_t OFF_H32   = 0;                                  // f32 16 MB
constexpr size_t OFF_HBF   = OFF_H32  + (size_t)cBS * cD * 4;    // bf16 8 MB
constexpr size_t OFF_QBF   = OFF_HBF  + (size_t)cBS * cD * 2;
constexpr size_t OFF_KBF   = OFF_QBF  + (size_t)cBS * cD * 2;
constexpr size_t OFF_VT    = OFF_KBF  + (size_t)cBS * cD * 2;    // V transposed [b,h,d,s]
constexpr size_t OFF_ABF   = OFF_VT   + (size_t)cBS * cD * 2;
constexpr size_t OFF_WQ    = OFF_ABF  + (size_t)cBS * cD * 2;
constexpr size_t OFF_WK    = OFF_WQ   + (size_t)cHD * cHD * 2;
constexpr size_t OFF_WV    = OFF_WK   + (size_t)cHD * cHD * 2;
constexpr size_t OFF_WO    = OFF_WV   + (size_t)cHD * cHD * 2;
constexpr size_t OFF_WIDTH = OFF_WO   + (size_t)cD * cD * 2;
// total ~ 59.3 MB

// ============ K1: embedding gather + sinusoidal position embedding ============
__global__ void k_embed(const int* __restrict__ x, const float* __restrict__ emb,
                        float* __restrict__ h32, __bf16* __restrict__ hbf) {
  int idx = blockIdx.x * blockDim.x + threadIdx.x;        // < B*S*D
  int d = idx & (cD - 1);
  int bs = idx / cD;
  int s = bs & (cS - 1);
  int tok = x[bs];
  int k2 = d & ~1;                                        // even pair index *2
  float div = __expf(-(9.210340371976184f / cD) * (float)k2);  // ln(10000)
  float ang = (float)s * div;
  float pe = (d & 1) ? __cosf(ang) : __sinf(ang);
  float v = emb[(size_t)tok * cD + d] + pe;
  h32[idx] = v;
  hbf[idx] = f2bf(v);
}

// ============ K2a: width[b,s] = h[b,s,:] . Wc + bc (wave per row) ============
__global__ void k_width(const float* __restrict__ h32, const float* __restrict__ Wc,
                        const float* __restrict__ bc, float* __restrict__ width) {
  int lane = threadIdx.x & 31;
  int wave = threadIdx.x >> 5;
  int row = blockIdx.x * (blockDim.x >> 5) + wave;        // < B*S
  const float* hr = h32 + (size_t)row * cD;
  float acc = 0.f;
  #pragma unroll
  for (int i = 0; i < cD / 32; ++i) acc += hr[lane + 32 * i] * Wc[lane + 32 * i];
  #pragma unroll
  for (int m = 16; m >= 1; m >>= 1) acc += __shfl_xor(acc, m, 32);
  if (lane == 0) width[row] = acc + bc[0];
}

// ============ K2b: per-batch min-max normalize over S ============
__global__ void k_norm(float* __restrict__ width) {
  __shared__ float smin[cS], smax[cS];
  int b = blockIdx.x, s = threadIdx.x;
  float w = width[b * cS + s];
  smin[s] = w; smax[s] = w;
  __syncthreads();
  for (int off = cS / 2; off >= 1; off >>= 1) {
    if (s < off) {
      smin[s] = fminf(smin[s], smin[s + off]);
      smax[s] = fmaxf(smax[s], smax[s + off]);
    }
    __syncthreads();
  }
  float lo = smin[0], hi = smax[0];
  width[b * cS + s] = (w - lo) / (hi - lo);
}

// ============ K3: converable mask, closed-form gather/move logic ============
__global__ void k_mask(const float* __restrict__ width, float* __restrict__ maskout) {
  int i = blockIdx.x & (cS - 1);
  int b = blockIdx.x / cS;
  float w = width[b * cS + i];
  int ip = (i - 512) & (cS - 1);                  // (i - shift) mod S
  int r = (cS - 1) - ip;
  bool lower = (i < 512);
  float* mrow = maskout + ((size_t)b * cS + i) * cS;
  for (int j = threadIdx.x; j < cS; j += blockDim.x) {
    int idx = (r + j) & (cS - 1);
    bool cmp = (idx >= j);
    bool move = lower ? cmp : !cmp;
    float origin = (idx < 512) ? (1.0f - (float)idx * (1.0f / 511.0f))
                               : ((float)(idx - 512) * (1.0f / 511.0f));
    float sg = 1.0f / (1.0f + __expf(-(origin - w) * 100.0f));
    mrow[j] = move ? (1.0f - sg) : 0.0f;
  }
}

// ============ K4: convert weights to bf16 ============
__global__ void k_wconv(const float* __restrict__ Wq, const float* __restrict__ Wk,
                        const float* __restrict__ Wv, const float* __restrict__ Wo,
                        __bf16* wq, __bf16* wk, __bf16* wv, __bf16* wo) {
  int i = blockIdx.x * blockDim.x + threadIdx.x;
  constexpr int HW = cHD * cHD;                   // 4096
  if (i < HW)               wq[i] = f2bf(Wq[i]);
  else if (i < 2 * HW)      wk[i - HW] = f2bf(Wk[i - HW]);
  else if (i < 3 * HW)      wv[i - 2 * HW] = f2bf(Wv[i - 2 * HW]);
  else if (i < 3 * HW + cD * cD) wo[i - 3 * HW] = f2bf(Wo[i - 3 * HW]);
}

// helper: 16(M) x 64(N) per-head projection, row-major out; LDS-staged b128 stores
__device__ __forceinline__ void proj_head_st(const v16bf& a0, const v16bf& a1,
                                             const __bf16* __restrict__ W,
                                             __bf16* __restrict__ O,
                                             __bf16 (*stile)[64],          // [16][64]
                                             int mbase, int head,
                                             int hl, int nl, int lane) {
  v8f acc[4];
  #pragma unroll
  for (int nt = 0; nt < 4; ++nt) {
    v16bf b0, b1;
    int o = nt * 16 + nl;
    #pragma unroll
    for (int e = 0; e < 16; ++e) {
      b0[e] = W[o * cHD + hl * 16 + e];
      b1[e] = W[o * cHD + 32 + hl * 16 + e];
    }
    v8f z = {};
    z = WMMA_BF16(a0, b0, z);
    acc[nt] = WMMA_BF16(a1, b1, z);
  }
  #pragma unroll
  for (int nt = 0; nt < 4; ++nt)
    #pragma unroll
    for (int rr = 0; rr < 8; ++rr)
      stile[rr + 8 * hl][nt * 16 + nl] = f2bf(acc[nt][rr]);   // lane-consecutive b16
  int r4 = lane >> 3, c8 = (lane & 7) * 8;
  #pragma unroll
  for (int i = 0; i < 4; ++i) {                               // 4 x b128 per lane
    int row = i * 4 + r4;
    *(v8bf*)(O + (size_t)(mbase + row) * cD + head * cHD + c8) =
        *(const v8bf*)(&stile[row][c8]);
  }
}

// helper: V projection, transposed out vT[b,h,d,s]; LDS-staged b128 stores
__device__ __forceinline__ void proj_head_T_st(const v16bf& a0, const v16bf& a1,
                                               const __bf16* __restrict__ W,
                                               __bf16* __restrict__ vT,
                                               __bf16 (*ttile)[24],        // [64][24]
                                               int mbase, int head,
                                               int hl, int nl, int lane) {
  int b = mbase / cS;
  int srow = mbase & (cS - 1);
  v8f acc[4];
  #pragma unroll
  for (int nt = 0; nt < 4; ++nt) {
    v16bf b0, b1;
    int o = nt * 16 + nl;
    #pragma unroll
    for (int e = 0; e < 16; ++e) {
      b0[e] = W[o * cHD + hl * 16 + e];
      b1[e] = W[o * cHD + 32 + hl * 16 + e];
    }
    v8f z = {};
    z = WMMA_BF16(a0, b0, z);
    acc[nt] = WMMA_BF16(a1, b1, z);
  }
  #pragma unroll
  for (int nt = 0; nt < 4; ++nt)
    #pragma unroll
    for (int rr = 0; rr < 8; ++rr)                            // 48B row stride: 16 banks
      ttile[nt * 16 + nl][rr + 8 * hl] = f2bf(acc[nt][rr]);
  __bf16* base = vT + (size_t)(b * cH + head) * cHD * cS + srow;
  int dd = lane >> 1, so = (lane & 1) * 8;
  #pragma unroll
  for (int i = 0; i < 4; ++i) {                               // 4 x b128 per lane
    int d = i * 16 + dd;
    *(v8bf*)(base + (size_t)d * cS + so) = *(const v8bf*)(&ttile[d][so]);
  }
}

// ============ K5: Q/K/V per-head projections (wave per 16-row M tile) ============
__global__ void k_qkv(const __bf16* __restrict__ hbf,
                      const __bf16* __restrict__ wq, const __bf16* __restrict__ wk,
                      const __bf16* __restrict__ wv,
                      __bf16* __restrict__ qbf, __bf16* __restrict__ kbf,
                      __bf16* __restrict__ vT) {
  __shared__ __bf16 stile[2][16][64];              // wave-private staging (4 KB)
  __shared__ __bf16 ttile[2][64][24];              // wave-private transposed (6 KB)
  int lane = threadIdx.x & 31;
  int wave = threadIdx.x >> 5;
  int mtile = blockIdx.x * (blockDim.x >> 5) + wave;
  int head = blockIdx.y;
  int mbase = mtile * 16;
  int hl = lane >> 4, nl = lane & 15;
  v16bf a0, a1;                                    // A: m = lane&15, contiguous k = 16*hl + e
  const __bf16* arow = hbf + (size_t)(mbase + nl) * cD + head * cHD;
  #pragma unroll
  for (int e = 0; e < 8; ++e) {
    a0[e]     = arow[hl * 16 + e];
    a0[8 + e] = arow[hl * 16 + 8 + e];
    a1[e]     = arow[32 + hl * 16 + e];
    a1[8 + e] = arow[32 + hl * 16 + 8 + e];
  }
  proj_head_st  (a0, a1, wq, qbf, stile[wave], mbase, head, hl, nl, lane);
  proj_head_st  (a0, a1, wk, kbf, stile[wave], mbase, head, hl, nl, lane);
  proj_head_T_st(a0, a1, wv, vT,  ttile[wave], mbase, head, hl, nl, lane);
}

// ============ K6: flash attention, wave per (b, h, 16-row Q tile) ============
// energy[b,h,q,k] = (Q.K) * mask[h,q,k] / sqrt(D)   (reference broadcast: B==H)
__global__ void __launch_bounds__(32) k_attn(const __bf16* __restrict__ qbf,
                                             const __bf16* __restrict__ kbf,
                                             const __bf16* __restrict__ vT,
                                             const float* __restrict__ mask,
                                             __bf16* __restrict__ abf) {
  __shared__ float  mstage[2][16][32];             // 2x2KB async mask double-buffer
  __shared__ __bf16 pstage[16 * 32];               // 1 KB C->A relayout stage
  __shared__ __bf16 otile[16][64];                 // 2 KB output staging
  int lane = threadIdx.x;
  int hl = lane >> 4, nl = lane & 15;
  int qtile = blockIdx.x, head = blockIdx.y, b = blockIdx.z;
  int qbase = qtile * 16;
  const float invs = 0.044194173824159216f;        // 1/sqrt(512)

  // Q A-fragments
  v16bf aq0, aq1;
  {
    const __bf16* qr = qbf + ((size_t)(b * cS) + qbase + nl) * cD + head * cHD;
    #pragma unroll
    for (int e = 0; e < 8; ++e) {
      aq0[e] = qr[hl * 16 + e];      aq0[8 + e] = qr[hl * 16 + 8 + e];
      aq1[e] = qr[32 + hl * 16 + e]; aq1[8 + e] = qr[32 + hl * 16 + 8 + e];
    }
  }

  const float* mrowbase = mask + ((size_t)head * cS + qbase) * cS;
  const __bf16* vhead = vT + (size_t)(b * cH + head) * cHD * cS;
  unsigned msbase = (unsigned)(size_t)(&mstage[0][0][0]);

  // async mask tile prefetch: 16 rows x 32 f32 = 2KB = 4 b128s per wave
  auto mask_prefetch = [&](int kbn, int buf) {
    int row = lane >> 3;                 // + 4 per instruction
    int col = (lane & 7) * 4;
    const float* g = mrowbase + (size_t)row * cS + kbn * 32 + col;
    unsigned l0 = msbase + (unsigned)buf * 2048u + (unsigned)lane * 16u;
    #pragma unroll
    for (int i = 0; i < 4; ++i) {
      asm volatile("global_load_async_to_lds_b128 %0, %1, off"
                   :: "v"(l0 + (unsigned)i * 512u), "v"(g) : "memory");
      g += 4 * cS;
    }
  };

  mask_prefetch(0, 0);                   // prime the pipeline

  float ml[8], ll[8];
  v8f o0 = {}, o1 = {}, o2 = {}, o3 = {};
  #pragma unroll
  for (int rr = 0; rr < 8; ++rr) { ml[rr] = -1e30f; ll[rr] = 0.f; }

  for (int kb = 0; kb < cS / 32; ++kb) {
    int kbase = kb * 32;
    int buf = kb & 1;
    mask_prefetch((kb + 1) & (cS / 32 - 1), buf ^ 1);   // overlap with QK^T

    // --- S = Q K^T : B[k=d][n=kvrow], two 16-col sub-tiles, K=64 in 2 chunks ---
    v16bf bk00, bk01, bk10, bk11;
    {
      const __bf16* k0 = kbf + ((size_t)(b * cS) + kbase + nl) * cD + head * cHD;
      const __bf16* k1 = k0 + (size_t)16 * cD;
      #pragma unroll
      for (int e = 0; e < 16; ++e) {
        bk00[e] = k0[hl * 16 + e];      bk01[e] = k0[32 + hl * 16 + e];
        bk10[e] = k1[hl * 16 + e];      bk11[e] = k1[32 + hl * 16 + e];
      }
    }
    v8f s0 = {}, s1 = {};
    s0 = WMMA_BF16(aq0, bk00, s0);  s0 = WMMA_BF16(aq1, bk01, s0);
    s1 = WMMA_BF16(aq0, bk10, s1);  s1 = WMMA_BF16(aq1, bk11, s1);

    // release current mask buffer (older 4 asyncs done; prefetch may still fly)
    asm volatile("s_wait_asynccnt 4" ::: "memory");

    // --- mask * scale, online softmax (rows live in 16-lane half-wave groups) ---
    float p0[8], p1[8];
    #pragma unroll
    for (int rr = 0; rr < 8; ++rr) {
      int row = rr + 8 * hl;
      float e0 = s0[rr] * mstage[buf][row][nl] * invs;
      float e1 = s1[rr] * mstage[buf][row][16 + nl] * invs;
      float mx = fmaxf(e0, e1);
      #pragma unroll
      for (int sh = 8; sh >= 1; sh >>= 1) mx = fmaxf(mx, __shfl_xor(mx, sh, 16));
      float mnew = fmaxf(ml[rr], mx);
      float c = __expf(ml[rr] - mnew);
      p0[rr] = __expf(e0 - mnew);
      p1[rr] = __expf(e1 - mnew);
      float rs = p0[rr] + p1[rr];
      #pragma unroll
      for (int sh = 8; sh >= 1; sh >>= 1) rs += __shfl_xor(rs, sh, 16);
      ll[rr] = ll[rr] * c + rs;
      ml[rr] = mnew;
      o0[rr] *= c; o1[rr] *= c; o2[rr] *= c; o3[rr] *= c;
    }

    // --- relayout P (C layout) -> A layout via LDS ---
    #pragma unroll
    for (int rr = 0; rr < 8; ++rr) {
      int row = rr + 8 * hl;
      pstage[row * 32 + nl]      = f2bf(p0[rr]);
      pstage[row * 32 + 16 + nl] = f2bf(p1[rr]);
    }
    __syncthreads();
    v16bf ap;
    #pragma unroll
    for (int e = 0; e < 8; ++e) {
      ap[e]     = pstage[nl * 32 + hl * 16 + e];
      ap[8 + e] = pstage[nl * 32 + hl * 16 + 8 + e];
    }
    __syncthreads();

    // --- O += P x V : V^T layout -> contiguous 16-bf16 per-lane B fragments ---
    {
      v16bf bv0, bv1, bv2, bv3;
      const __bf16* vc0 = vhead + (size_t)(nl)      * cS + kbase + hl * 16;
      const __bf16* vc1 = vhead + (size_t)(16 + nl) * cS + kbase + hl * 16;
      const __bf16* vc2 = vhead + (size_t)(32 + nl) * cS + kbase + hl * 16;
      const __bf16* vc3 = vhead + (size_t)(48 + nl) * cS + kbase + hl * 16;
      #pragma unroll
      for (int e = 0; e < 16; ++e) {
        bv0[e] = vc0[e]; bv1[e] = vc1[e]; bv2[e] = vc2[e]; bv3[e] = vc3[e];
      }
      o0 = WMMA_BF16(ap, bv0, o0);
      o1 = WMMA_BF16(ap, bv1, o1);
      o2 = WMMA_BF16(ap, bv2, o2);
      o3 = WMMA_BF16(ap, bv3, o3);
    }
  }

  // --- epilogue: normalize, stage 16x64 tile in LDS, coalesced b128 stores ---
  #pragma unroll
  for (int rr = 0; rr < 8; ++rr) {
    float inv = 1.0f / ll[rr];
    int row = rr + 8 * hl;
    otile[row][nl]      = f2bf(o0[rr] * inv);
    otile[row][16 + nl] = f2bf(o1[rr] * inv);
    otile[row][32 + nl] = f2bf(o2[rr] * inv);
    otile[row][48 + nl] = f2bf(o3[rr] * inv);
  }
  int r4 = lane >> 3, c8 = (lane & 7) * 8;
  #pragma unroll
  for (int i = 0; i < 4; ++i) {
    int row = i * 4 + r4;
    *(v8bf*)(abf + ((size_t)(b * cS) + qbase + row) * cD + head * cHD + c8) =
        *(const v8bf*)(&otile[row][c8]);
  }
}

// ============ K7: out = attn @ Wo^T + bo  (M=8192, N=512, K=512) ============
__global__ void k_oproj(const __bf16* __restrict__ abf, const __bf16* __restrict__ wobf,
                        const float* __restrict__ bo, float* __restrict__ out) {
  int lane = threadIdx.x & 31;
  int wave = threadIdx.x >> 5;
  int hl = lane >> 4, nl = lane & 15;
  int mtile = blockIdx.x;
  int ntile = blockIdx.y * (blockDim.x >> 5) + wave;
  int mbase = mtile * 16, nbase = ntile * 16;
  v8f acc = {};
  const __bf16* arow = abf + (size_t)(mbase + nl) * cD;
  const __bf16* wrow = wobf + (size_t)(nbase + nl) * cD;   // B[k=d][n=o] = Wo[o,d]
  for (int kc = 0; kc < cD / 32; ++kc) {
    v16bf a, bm;
    int ab = kc * 32;
    #pragma unroll
    for (int e = 0; e < 8; ++e) {
      a[e]     = arow[ab + hl * 16 + e];
      a[8 + e] = arow[ab + hl * 16 + 8 + e];
    }
    #pragma unroll
    for (int e = 0; e < 16; ++e) bm[e] = wrow[ab + hl * 16 + e];
    acc = WMMA_BF16(a, bm, acc);
  }
  float bias = bo[nbase + nl];
  #pragma unroll
  for (int rr = 0; rr < 8; ++rr)
    out[(size_t)(mbase + rr + 8 * hl) * cD + nbase + nl] = acc[rr] + bias;
}

// ================================ launcher ================================
extern "C" void kernel_launch(void* const* d_in, const int* in_sizes, int n_in,
                              void* d_out, int out_size, void* d_ws, size_t ws_size,
                              hipStream_t stream) {
  const int*   x   = (const int*)  d_in[0];
  const float* emb = (const float*)d_in[1];
  const float* Wq  = (const float*)d_in[2];
  const float* Wk  = (const float*)d_in[3];
  const float* Wv  = (const float*)d_in[4];
  const float* Wo  = (const float*)d_in[5];
  const float* bo  = (const float*)d_in[6];
  const float* Wc  = (const float*)d_in[7];
  const float* bc  = (const float*)d_in[8];

  float* out     = (float*)d_out;
  float* maskout = out + OUT0;

  char* ws = (char*)d_ws;
  float*  h32   = (float*) (ws + OFF_H32);
  __bf16* hbf   = (__bf16*)(ws + OFF_HBF);
  __bf16* qbf   = (__bf16*)(ws + OFF_QBF);
  __bf16* kbf   = (__bf16*)(ws + OFF_KBF);
  __bf16* vT    = (__bf16*)(ws + OFF_VT);
  __bf16* abf   = (__bf16*)(ws + OFF_ABF);
  __bf16* wqb   = (__bf16*)(ws + OFF_WQ);
  __bf16* wkb   = (__bf16*)(ws + OFF_WK);
  __bf16* wvb   = (__bf16*)(ws + OFF_WV);
  __bf16* wob   = (__bf16*)(ws + OFF_WO);
  float*  width = (float*) (ws + OFF_WIDTH);

  k_embed<<<dim3(cBS * cD / 256), 256, 0, stream>>>(x, emb, h32, hbf);
  k_width<<<dim3(cBS / 8), 256, 0, stream>>>(h32, Wc, bc, width);
  k_norm<<<dim3(cB), cS, 0, stream>>>(width);
  k_mask<<<dim3(cB * cS), 256, 0, stream>>>(width, maskout);
  k_wconv<<<dim3((3 * cHD * cHD + cD * cD + 255) / 256), 256, 0, stream>>>(
      Wq, Wk, Wv, Wo, wqb, wkb, wvb, wob);
  k_qkv<<<dim3(cBS / 16 / 2, cH), 64, 0, stream>>>(hbf, wqb, wkb, wvb, qbf, kbf, vT);
  k_attn<<<dim3(cS / 16, cH, cB), 32, 0, stream>>>(qbf, kbf, vT, maskout, abf);
  k_oproj<<<dim3(cBS / 16, cD / 16 / 4), 128, 0, stream>>>(abf, wob, bo, out);
}
